// SelfAttention_30554397343752
// MI455X (gfx1250) — matrix operand order repaired and checked by reading, not compiled, not run on hardware
//
#include <hip/hip_runtime.h>
#include <hip/hip_bf16.h>

// Problem constants (from reference)
#define B_    2
#define S_    2048
#define D_    1024
#define H_    16
#define HD_   64
#define HALF_ 32
#define M_    (B_ * S_)   // 4096 rows
#define EPS_  1e-6f

typedef __attribute__((ext_vector_type(16))) __bf16 v16bf;
typedef __attribute__((ext_vector_type(8)))  __bf16 v8bf;
typedef __attribute__((ext_vector_type(4)))  __bf16 v4bf;
typedef __attribute__((ext_vector_type(8)))  float  v8f;

// Assemble a 16x32 A-fragment (or 32x16 B-fragment) from two 16-byte chunks.
static __device__ __forceinline__ v16bf make_frag(v8bf lo, v8bf hi) {
  v16bf r;
#pragma unroll
  for (int i = 0; i < 8; ++i) { r[i] = lo[i]; r[i + 8] = hi[i]; }
  return r;
}

static __device__ __forceinline__ v8f wmma_bf16(v16bf a, v16bf b, v8f c) {
  // 8 args: (neg_a, A, neg_b, B, c_mod, C, reuse_a, reuse_b)
  return __builtin_amdgcn_wmma_f32_16x16x32_bf16(false, a, false, b,
                                                 (short)0, c, false, false);
}

// Async global -> LDS copy, 16 bytes per lane. Tracked by ASYNCcnt.
// (CDNA5 ISA: GLOBAL_LOAD_ASYNC_TO_LDS_B128, GV mode)
static __device__ __forceinline__ void async_load_b128(unsigned lds_off,
                                                       const void* gptr) {
  asm volatile("global_load_async_to_lds_b128 %0, %1, off"
               :: "v"(lds_off), "v"((unsigned long long)(size_t)gptr)
               : "memory");
}
static __device__ __forceinline__ void wait_asynccnt0() {
  asm volatile("s_wait_asynccnt 0x0" ::: "memory");
}

// ---------------------------------------------------------------------------
// GEMM: C[m,n] = sum_k A[m,k] * W[n,k] + bias[n]
// M=4096, N=1024, K=1024, fp32 in, bf16 WMMA, fp32 out.
// WG tile 128x128, K-step 64, 8 waves -> each wave does a 32x64 sub-tile.
// 16 WMMAs per barrier pair per wave.
// ---------------------------------------------------------------------------
__global__ void __launch_bounds__(256)
gemm_xwt_bias(const float* __restrict__ A, const float* __restrict__ W,
              const float* __restrict__ bias, float* __restrict__ C) {
  __shared__ __align__(64) __bf16 As[128][72];  // +8 pad: conflict-free b128 ds loads
  __shared__ __align__(64) __bf16 Bs[128][72];

  const int tid   = threadIdx.x;
  const int lane  = tid & 31;
  const int wid   = tid >> 5;
  const int ln    = lane & 15;   // M/N index within a 16-tile
  const int hf    = lane >> 4;   // selects K sub-range per documented layout
  const int waveM = wid & 3;     // 4 waves along M
  const int waveN = wid >> 2;    // 2 waves along N
  const int mbase = blockIdx.y * 128;
  const int nbase = blockIdx.x * 128;

  v8f acc[2][4];
#pragma unroll
  for (int mt = 0; mt < 2; ++mt)
#pragma unroll
    for (int nt = 0; nt < 4; ++nt)
#pragma unroll
      for (int r = 0; r < 8; ++r) acc[mt][nt][r] = 0.f;

  for (int k0 = 0; k0 < 1024; k0 += 64) {
    // Global fp32 -> bf16 LDS tiles (8192 elems each; 256 thr x 8 reps x 4)
#pragma unroll
    for (int rep = 0; rep < 8; ++rep) {
      int idx = rep * 1024 + tid * 4;
      int row = idx >> 6, col = idx & 63;
      float4 va = *(const float4*)&A[(size_t)(mbase + row) * 1024 + k0 + col];
      v4bf ta; ta[0] = (__bf16)va.x; ta[1] = (__bf16)va.y;
               ta[2] = (__bf16)va.z; ta[3] = (__bf16)va.w;
      *(v4bf*)&As[row][col] = ta;
      float4 vb = *(const float4*)&W[(size_t)(nbase + row) * 1024 + k0 + col];
      v4bf tb; tb[0] = (__bf16)vb.x; tb[1] = (__bf16)vb.y;
               tb[2] = (__bf16)vb.z; tb[3] = (__bf16)vb.w;
      *(v4bf*)&Bs[row][col] = tb;
    }
    __syncthreads();

#pragma unroll
    for (int ks = 0; ks < 2; ++ks) {
      v16bf af[2], bfr[4];
#pragma unroll
      for (int mt = 0; mt < 2; ++mt) {
        int row = waveM * 32 + mt * 16 + ln;
        af[mt] = make_frag(*(const v8bf*)&As[row][ks * 32 + hf * 8],
                           *(const v8bf*)&As[row][ks * 32 + 16 + hf * 8]);
      }
#pragma unroll
      for (int nt = 0; nt < 4; ++nt) {
        int col = waveN * 64 + nt * 16 + ln;      // B col n = W row n
        bfr[nt] = make_frag(*(const v8bf*)&Bs[col][ks * 32 + hf * 8],
                            *(const v8bf*)&Bs[col][ks * 32 + 16 + hf * 8]);
      }
#pragma unroll
      for (int mt = 0; mt < 2; ++mt)
#pragma unroll
        for (int nt = 0; nt < 4; ++nt)
          acc[mt][nt] = wmma_bf16(af[mt], bfr[nt], acc[mt][nt]);
    }
    __syncthreads();
  }

  // Epilogue: C fragment row m = r + 8*hf, col n = ln (per 16x16 tile)
#pragma unroll
  for (int nt = 0; nt < 4; ++nt) {
    int n = nbase + waveN * 64 + nt * 16 + ln;
    float bb = bias[n];
#pragma unroll
    for (int mt = 0; mt < 2; ++mt) {
      int m0 = mbase + waveM * 32 + mt * 16 + hf * 8;
#pragma unroll
      for (int r = 0; r < 8; ++r)
        C[(size_t)(m0 + r) * 1024 + n] = acc[mt][nt][r] + bb;
    }
  }
}

// ---------------------------------------------------------------------------
// RMSNorm (q,k) + RoPE (q,k) + repack into bf16 head-major layouts:
//   Qh,Kh: [B,H,S,HD]   Vt: [B,H,HD,S]  (transposed so PV B-operand is rowwise)
// One 256-thread block per token row.
// ---------------------------------------------------------------------------
__global__ void __launch_bounds__(256)
norm_rope_pack(const float* __restrict__ qraw, const float* __restrict__ kraw,
               const float* __restrict__ vraw,
               const float* __restrict__ gq, const float* __restrict__ gk,
               const float* __restrict__ fcos, const float* __restrict__ fsin,
               __bf16* __restrict__ Qh, __bf16* __restrict__ Kh,
               __bf16* __restrict__ Vt) {
  const int row = blockIdx.x;            // b*S + s
  const int b = row / S_, s = row % S_;
  const int tid = threadIdx.x;
  const float* qr = qraw + (size_t)row * D_;
  const float* kr = kraw + (size_t)row * D_;
  const float* vr = vraw + (size_t)row * D_;

  __shared__ float redq[256], redk[256];
  float sq = 0.f, sk = 0.f;
#pragma unroll
  for (int j = tid; j < D_; j += 256) {
    float a = qr[j]; sq += a * a;
    float c = kr[j]; sk += c * c;
  }
  redq[tid] = sq; redk[tid] = sk;
  __syncthreads();
  for (int off = 128; off > 0; off >>= 1) {
    if (tid < off) { redq[tid] += redq[tid + off]; redk[tid] += redk[tid + off]; }
    __syncthreads();
  }
  const float rq = rsqrtf(redq[0] * (1.f / D_) + EPS_);
  const float rk = rsqrtf(redk[0] * (1.f / D_) + EPS_);

  // 512 rotary pairs per row; interleaved (even,odd) convention
#pragma unroll
  for (int pp = 0; pp < 2; ++pp) {
    int p  = tid + pp * 256;
    int h  = p >> 5;                      // p / HALF_
    int j  = p & 31;
    int d0 = h * HD_ + 2 * j, d1 = d0 + 1;
    float c  = fcos[s * HALF_ + j];
    float sn = fsin[s * HALF_ + j];
    size_t ob = ((size_t)(b * H_ + h) * S_ + s) * HD_ + 2 * j;
    float q0 = qr[d0] * rq * gq[d0], q1 = qr[d1] * rq * gq[d1];
    Qh[ob]     = (__bf16)(q0 * c - q1 * sn);
    Qh[ob + 1] = (__bf16)(q0 * sn + q1 * c);
    float k0 = kr[d0] * rk * gk[d0], k1 = kr[d1] * rk * gk[d1];
    Kh[ob]     = (__bf16)(k0 * c - k1 * sn);
    Kh[ob + 1] = (__bf16)(k0 * sn + k1 * c);
  }
  // V: no norm/rope; store transposed per head
#pragma unroll
  for (int dd = tid; dd < D_; dd += 256) {
    int h = dd >> 6, d = dd & 63;
    Vt[((size_t)(b * H_ + h) * HD_ + d) * S_ + s] = (__bf16)vr[dd];
  }
}

// ---------------------------------------------------------------------------
// Flash attention: grid (S/64 q-blocks, B*H heads), 128 threads = 4 waves.
// Each wave owns a 16-row Q tile; loop over 64-key blocks with online softmax.
// K/V tiles staged with GLOBAL_LOAD_ASYNC_TO_LDS_B128 (ASYNCcnt).
// ---------------------------------------------------------------------------
__global__ void __launch_bounds__(128)
flash_attn(const __bf16* __restrict__ Qh, const __bf16* __restrict__ Kh,
           const __bf16* __restrict__ Vt, float* __restrict__ O) {
  __shared__ __align__(64) __bf16 Ks[64][72];      // [key][d], +8 pad
  __shared__ __align__(64) __bf16 Vs[64][72];      // [d][key]
  __shared__ __align__(64) __bf16 Ps[4][16][72];   // per-wave P tile (C->A relayout)

  const int tid  = threadIdx.x;
  const int lane = tid & 31;
  const int wid  = tid >> 5;
  const int ln   = lane & 15;
  const int hf   = lane >> 4;
  const int bh   = blockIdx.y;           // b*H + h
  const int b    = bh / H_, h = bh % H_;
  const int qbase = blockIdx.x * 64;
  const size_t soff = (size_t)bh * S_;   // row base in Qh/Kh
  const size_t voff = (size_t)bh * HD_;  // row base in Vt

  // Q fragments live in registers across the whole key loop
  const __bf16* qptr = Qh + (soff + qbase + wid * 16 + ln) * HD_;
  v16bf qf[2];
#pragma unroll
  for (int ks = 0; ks < 2; ++ks)
    qf[ks] = make_frag(*(const v8bf*)&qptr[ks * 32 + hf * 8],
                       *(const v8bf*)&qptr[ks * 32 + 16 + hf * 8]);

  v8f oacc[4];
  float mrow[8], lrow[8];
#pragma unroll
  for (int nt = 0; nt < 4; ++nt)
#pragma unroll
    for (int r = 0; r < 8; ++r) oacc[nt][r] = 0.f;
#pragma unroll
  for (int r = 0; r < 8; ++r) { mrow[r] = -1e30f; lrow[r] = 0.f; }

  for (int kb = 0; kb < S_; kb += 64) {
    __syncthreads();
    // Async K/V tile copy to LDS (4096 bf16 each; 128 thr x 4 reps x 16B)
#pragma unroll
    for (int rep = 0; rep < 4; ++rep) {
      int idx = rep * 1024 + tid * 8;
      int r0 = idx >> 6, c0 = idx & 63;
      async_load_b128((unsigned)(size_t)&Ks[r0][c0],
                      &Kh[(soff + kb + r0) * HD_ + c0]);
      async_load_b128((unsigned)(size_t)&Vs[r0][c0],
                      &Vt[(voff + r0) * S_ + kb + c0]);
    }
    wait_asynccnt0();
    __syncthreads();

    // Prefetch next K/V tiles into cache while we compute on this one
    if (kb + 64 < S_) {
      if (tid < 64)
        __builtin_prefetch(&Kh[(soff + kb + 64 + tid) * HD_], 0, 0);
      else
        __builtin_prefetch(&Vt[(voff + (tid - 64)) * S_ + kb + 64], 0, 0);
    }

    // S = Q K^T  (16 q-rows x 64 keys), k-dim = HD = 2 x 32
    v8f sf[4];
#pragma unroll
    for (int nt = 0; nt < 4; ++nt) {
#pragma unroll
      for (int r = 0; r < 8; ++r) sf[nt][r] = 0.f;
#pragma unroll
      for (int ks = 0; ks < 2; ++ks) {
        v16bf kf = make_frag(*(const v8bf*)&Ks[nt * 16 + ln][ks * 32 + hf * 8],
                             *(const v8bf*)&Ks[nt * 16 + ln][ks * 32 + 16 + hf * 8]);
        sf[nt] = wmma_bf16(qf[ks], kf, sf[nt]);
      }
    }

    // Online softmax (scale 1/sqrt(HD) = 0.125); rows live across 16 lanes of a half
    float corr[8];
#pragma unroll
    for (int r = 0; r < 8; ++r) {
      float mx = -1e30f;
#pragma unroll
      for (int nt = 0; nt < 4; ++nt) mx = fmaxf(mx, sf[nt][r]);
      mx *= 0.125f;
      mx = fmaxf(mx, __shfl_xor(mx, 1));
      mx = fmaxf(mx, __shfl_xor(mx, 2));
      mx = fmaxf(mx, __shfl_xor(mx, 4));
      mx = fmaxf(mx, __shfl_xor(mx, 8));
      float nm = fmaxf(mrow[r], mx);
      corr[r] = __expf(mrow[r] - nm);
      mrow[r] = nm;
      float rs = 0.f;
#pragma unroll
      for (int nt = 0; nt < 4; ++nt) {
        float p = __expf(sf[nt][r] * 0.125f - nm);
        sf[nt][r] = p;
        rs += p;
      }
      rs += __shfl_xor(rs, 1);
      rs += __shfl_xor(rs, 2);
      rs += __shfl_xor(rs, 4);
      rs += __shfl_xor(rs, 8);
      lrow[r] = lrow[r] * corr[r] + rs;
#pragma unroll
      for (int nt = 0; nt < 4; ++nt) oacc[nt][r] *= corr[r];
    }

    // P: C-layout -> LDS -> A-layout (per-wave region; same-wave DS ops are in order)
#pragma unroll
    for (int nt = 0; nt < 4; ++nt)
#pragma unroll
      for (int r = 0; r < 8; ++r)
        Ps[wid][r + 8 * hf][nt * 16 + ln] = (__bf16)sf[nt][r];
    __builtin_amdgcn_wave_barrier();
    v16bf pf[2];
#pragma unroll
    for (int ks = 0; ks < 2; ++ks)
      pf[ks] = make_frag(*(const v8bf*)&Ps[wid][ln][ks * 32 + hf * 8],
                         *(const v8bf*)&Ps[wid][ln][ks * 32 + 16 + hf * 8]);

    // O += P V   (k-dim = 64 keys = 2 x 32; B[kk,d] = Vs[d][kk])
#pragma unroll
    for (int nt = 0; nt < 4; ++nt)
#pragma unroll
      for (int ks = 0; ks < 2; ++ks) {
        v16bf vf = make_frag(*(const v8bf*)&Vs[nt * 16 + ln][ks * 32 + hf * 8],
                             *(const v8bf*)&Vs[nt * 16 + ln][ks * 32 + 16 + hf * 8]);
        oacc[nt] = wmma_bf16(pf[ks], vf, oacc[nt]);
      }
  }

  // Normalize and write fp32 [B,S,D]
#pragma unroll
  for (int nt = 0; nt < 4; ++nt)
#pragma unroll
    for (int r = 0; r < 8; ++r) {
      int srow = qbase + wid * 16 + r + 8 * hf;
      O[((size_t)b * S_ + srow) * D_ + h * HD_ + nt * 16 + ln] =
          oacc[nt][r] / lrow[r];
    }
}

// ---------------------------------------------------------------------------
extern "C" void kernel_launch(void* const* d_in, const int* in_sizes, int n_in,
                              void* d_out, int out_size, void* d_ws, size_t ws_size,
                              hipStream_t stream) {
  (void)in_sizes; (void)n_in; (void)out_size; (void)ws_size;
  const float* x    = (const float*)d_in[0];
  const float* Wq   = (const float*)d_in[1];
  const float* bq   = (const float*)d_in[2];
  const float* Wk   = (const float*)d_in[3];
  const float* bk   = (const float*)d_in[4];
  const float* Wv   = (const float*)d_in[5];
  const float* bv   = (const float*)d_in[6];
  const float* Wo   = (const float*)d_in[7];
  const float* bo   = (const float*)d_in[8];
  const float* gq   = (const float*)d_in[9];
  const float* gk   = (const float*)d_in[10];
  const float* fcos = (const float*)d_in[11];
  const float* fsin = (const float*)d_in[12];

  // Workspace layout (fp32 then bf16): 4*16MB + 3*8MB = 88MB
  float* qraw = (float*)d_ws;
  float* kraw = qraw + (size_t)M_ * D_;
  float* vraw = kraw + (size_t)M_ * D_;
  float* attn = vraw + (size_t)M_ * D_;
  __bf16* Qh  = (__bf16*)(attn + (size_t)M_ * D_);
  __bf16* Kh  = Qh + (size_t)M_ * D_;
  __bf16* Vt  = Kh + (size_t)M_ * D_;

  dim3 ggrid(D_ / 128, M_ / 128);                 // (8, 32)
  gemm_xwt_bias<<<ggrid, 256, 0, stream>>>(x, Wq, bq, qraw);
  gemm_xwt_bias<<<ggrid, 256, 0, stream>>>(x, Wk, bk, kraw);
  gemm_xwt_bias<<<ggrid, 256, 0, stream>>>(x, Wv, bv, vraw);

  norm_rope_pack<<<M_, 256, 0, stream>>>(qraw, kraw, vraw, gq, gk, fcos, fsin,
                                         Qh, Kh, Vt);

  flash_attn<<<dim3(S_ / 64, B_ * H_), 128, 0, stream>>>(Qh, Kh, Vt, attn);

  gemm_xwt_bias<<<ggrid, 256, 0, stream>>>(attn, Wo, bo, (float*)d_out);
}
